// SLSTMCell_83640193122895
// MI455X (gfx1250) — matrix operand order repaired and checked by reading, not compiled, not run on hardware
//
#include <hip/hip_runtime.h>
#include <hip/hip_bf16.h>

// ---------------------------------------------------------------------------
// sLSTM cell, fused: one bf16-WMMA GEMM (M=B*L=65536, K=6H=1536, N=6H=1536)
// + in-register gate epilogue.
//
// Roofline: 2*M*K*N ~= 309 GFLOP vs ~460 MB HBM traffic -> compute-bound, so
// use v_wmma_f32_16x16x32_bf16 (CDNA5's high-rate matrix path for f32 data),
// f32 accumulation.  Weights pre-packed once to bf16 B-fragment layout in
// d_ws (4.5 MB, L2-resident).  M-tile = 32 rows/block so each B fragment
// fetched from L2 feeds two WMMAs (L2 weight traffic: 9.2 GB total, below
// the ~124us bf16 compute time).  16 waves/block; each wave owns one 16-wide
// channel tile across all 6 gates x 2 M-subtiles = 12 f32 accumulators, so
// the sigmoid/softmax/tanh epilogue never leaves registers.
// ---------------------------------------------------------------------------

#define B_  32
#define L_  2048
#define H_  256
#define K_  1536          // 6*H concat width
#define KB_ 48            // K_/32 K-steps per WMMA chain
#define NT_ 96            // N/16 = 6 gates * 16 channel-tiles
#define MT_ 32            // rows per block (2 WMMA M-subtiles)
#define WPACK_DWORDS (NT_ * KB_ * 32 * 8)   // 1,179,648 dwords = 4.5 MiB
#define ALDS_DWORDS  (2 * KB_ * 32 * 8)     // 24,576 dwords = 96 KiB (dynamic)

typedef __attribute__((ext_vector_type(16))) __bf16        v16bf;
typedef __attribute__((ext_vector_type(8)))  float         v8f;
typedef __attribute__((ext_vector_type(8)))  unsigned int  v8u;

__device__ __forceinline__ unsigned pack_bf16x2(float a, float b) {
    unsigned short ua = __builtin_bit_cast(unsigned short, (__bf16)a);
    unsigned short ub = __builtin_bit_cast(unsigned short, (__bf16)b);
    return (unsigned)ua | ((unsigned)ub << 16);
}

__device__ __forceinline__ float sigm(float v) {
    return 1.0f / (1.0f + __expf(-v));
}

// ---------------------------------------------------------------------------
// Pre-pack Ws [6,1536,256] f32 -> bf16 dwords in WMMA B-fragment order:
//   wp[((nt*KB_ + kb)*32 + lane)*8 + reg]
// B order used here (contiguous k, not AMD's table order): lane = n + 16*(K>=16), reg = (K&15)>>1,
// dword packs (K_even, K_odd).
// ---------------------------------------------------------------------------
__global__ void slstm_pack_w(const float* __restrict__ Ws,
                             unsigned* __restrict__ wp) {
    int flat = blockIdx.x * 256 + threadIdx.x;
    if (flat >= WPACK_DWORDS) return;
    int reg  = flat & 7;
    int idx  = flat >> 3;
    int lane = idx & 31;
    int t2   = idx >> 5;
    int kb   = t2 % KB_;
    int nt   = t2 / KB_;                 // 0..95
    int g    = nt >> 4;                  // gate 0..5
    int n    = ((nt & 15) << 4) + (lane & 15);
    int p    = reg + ((lane >> 4) << 3); // K-pair index 0..15
    int k0   = kb * 32 + 2 * p;
    float w0 = Ws[((size_t)(g * K_ + k0)     ) * H_ + n];
    float w1 = Ws[((size_t)(g * K_ + k0 + 1) ) * H_ + n];
    wp[flat] = pack_bf16x2(w0, w1);
}

// ---------------------------------------------------------------------------
// Main fused kernel: 1 block = 32 rows of (b,l), 512 threads = 16 waves.
// Wave w owns channels [w*16, w*16+16) for ALL six gates and both M-subtiles.
// ---------------------------------------------------------------------------
__global__ __launch_bounds__(512) void slstm_main(
        const float* __restrict__ h_t,  const float* __restrict__ x,
        const float* __restrict__ h_sl, const float* __restrict__ h_in,
        const float* __restrict__ c_t,  const float* __restrict__ bs,
        const unsigned* __restrict__ wp,
        float* __restrict__ outH, float* __restrict__ outC) {

    // A tile (32 rows x 1536 K) in bf16, WMMA A-fragment layout:
    //   Alds[((mt*KB_ + kb)*32 + lane)*8 + reg]  ; 96 KB dynamic LDS.
    extern __shared__ __align__(64) unsigned Alds[];

    const int tid  = threadIdx.x;
    const int lane = tid & 31;
    const int wave = tid >> 5;                 // 0..15 (= channel tile)
    const long r0  = (long)blockIdx.x * MT_;   // 32 | 2048 -> no batch crossing

    // ---- Stage concat tile into LDS (fp32 -> bf16, fragment order) --------
    // A order used here (contiguous k, matches B pack): lane = m + 16*h, reg r holds K-pair
    // p = r + (h<<3), dword packs (K_even, K_odd).
    for (int it = 0; it < ALDS_DWORDS / 512; ++it) {
        int flat = it * 512 + tid;             // 24576 dwords total
        int mt   = flat / 12288;               // M-subtile 0/1 (12288 dwords each)
        int rem  = flat % 12288;
        int reg  = rem & 7;
        int sl   = (rem >> 3) & 31;
        int kb   = rem >> 8;                   // 0..47
        int m    = (mt << 4) + (sl & 15);
        int p    = reg + ((sl >> 4) << 3);
        int seg  = kb >> 3;                    // which of the 6 concat pieces
        int ch   = ((kb & 7) << 5) + 2 * p;    // channel within piece (0..254)
        long r   = r0 + m;
        int  l   = (int)(r & (L_ - 1));
        const float* src = nullptr;
        switch (seg) {
            case 0: src = h_t  + r * H_;                                    break;
            case 1: src = (l > 0)      ? (h_t + (r - 1) * H_) : nullptr;    break;
            case 2: src = (l < L_ - 1) ? (h_t + (r + 1) * H_) : nullptr;    break;
            case 3: src = x    + r * H_;                                    break;
            case 4: src = h_sl + r * H_;                                    break;
            case 5: src = h_in + r * H_;                                    break;
        }
        float a0 = 0.0f, a1 = 0.0f;
        if (src) { a0 = src[ch]; a1 = src[ch + 1]; }
        Alds[flat] = pack_bf16x2(a0, a1);
    }
    __syncthreads();

    // ---- GEMM: 12 accumulators = 6 gates x 2 M-subtiles -------------------
    v8f acc[12] = {};
    for (int kb = 0; kb < KB_; ++kb) {
        // shared A fragments for this K-step (2x ds_load_b128 each)
        v16bf a0 = __builtin_bit_cast(
            v16bf, *(const v8u*)&Alds[((0 * KB_ + kb) * 32 + lane) * 8]);
        v16bf a1 = __builtin_bit_cast(
            v16bf, *(const v8u*)&Alds[((1 * KB_ + kb) * 32 + lane) * 8]);

        if (kb + 1 < KB_) {   // warm next K-step's weight stream (L2-resident)
            __builtin_prefetch(
                &wp[(((size_t)wave * KB_ + kb + 1) * 32 + lane) * 8], 0, 3);
        }
#pragma unroll
        for (int g = 0; g < 6; ++g) {
            int nt = g * 16 + wave;
            v16bf b = __builtin_bit_cast(
                v16bf, *(const v8u*)&wp[(((size_t)nt * KB_ + kb) * 32 + lane) * 8]);
            acc[g * 2 + 0] = __builtin_amdgcn_wmma_f32_16x16x32_bf16(
                false, a0, false, b, (short)0, acc[g * 2 + 0], false, false);
            acc[g * 2 + 1] = __builtin_amdgcn_wmma_f32_16x16x32_bf16(
                false, a1, false, b, (short)0, acc[g * 2 + 1], false, false);
        }
    }

    // ---- Epilogue (C layout: lane = n + 16*(m>=8), reg j -> m = j + 8*hi) -
    const int nlo = lane & 15;
    const int hi  = lane >> 4;
    const int ch  = wave * 16 + nlo;
    float bias[6];
#pragma unroll
    for (int g = 0; g < 6; ++g) bias[g] = bs[g * H_ + ch];

#pragma unroll
    for (int mt = 0; mt < 2; ++mt) {
#pragma unroll
        for (int j = 0; j < 8; ++j) {
            int  m = (mt << 4) + j + (hi << 3);
            long r = r0 + m;
            int  l = (int)(r & (L_ - 1));
            float i_t = sigm(acc[0 * 2 + mt][j] + bias[0]);
            float o_t = sigm(acc[1 * 2 + mt][j] + bias[1]);
            float f_t = sigm(acc[2 * 2 + mt][j] + bias[2]);
            float l_t = sigm(acc[3 * 2 + mt][j] + bias[3]);
            float r_t = sigm(acc[4 * 2 + mt][j] + bias[4]);
            float u_t = tanhf(acc[5 * 2 + mt][j] + bias[5]);
            // softmax over the (already sigmoided) {i,f,l,r}
            float mx = fmaxf(fmaxf(i_t, f_t), fmaxf(l_t, r_t));
            float ei = __expf(i_t - mx), ef = __expf(f_t - mx);
            float el = __expf(l_t - mx), er = __expf(r_t - mx);
            float inv = 1.0f / (ei + ef + el + er);
            i_t = ei * inv; f_t = ef * inv; l_t = el * inv; r_t = er * inv;

            float cm = c_t[r * H_ + ch];
            float cl = (l > 0)      ? c_t[(r - 1) * H_ + ch] : 0.0f;
            float cr = (l < L_ - 1) ? c_t[(r + 1) * H_ + ch] : 0.0f;
            float cnew = f_t * cm + l_t * cl + r_t * cr + i_t * u_t;
            float hnew = o_t * tanhf(cnew);
            outH[r * H_ + ch] = hnew;
            outC[r * H_ + ch] = cnew;
        }
    }
}

// ---------------------------------------------------------------------------
extern "C" void kernel_launch(void* const* d_in, const int* in_sizes, int n_in,
                              void* d_out, int out_size, void* d_ws, size_t ws_size,
                              hipStream_t stream) {
    (void)in_sizes; (void)n_in; (void)out_size; (void)ws_size;
    const float* h_t  = (const float*)d_in[0];
    const float* x    = (const float*)d_in[1];
    const float* h_sl = (const float*)d_in[2];
    const float* h_in = (const float*)d_in[3];
    const float* c_t  = (const float*)d_in[4];
    const float* Ws   = (const float*)d_in[5];
    const float* bs   = (const float*)d_in[6];

    float* outH = (float*)d_out;
    float* outC = outH + (size_t)B_ * L_ * H_;
    unsigned* wp = (unsigned*)d_ws;   // 4.5 MiB packed bf16 weights

    slstm_pack_w<<<(WPACK_DWORDS + 255) / 256, 256, 0, stream>>>(Ws, wp);

    int nblocks = (B_ * L_) / MT_;    // 2048 blocks x 512 threads (16 waves)
    size_t lds_bytes = (size_t)ALDS_DWORDS * sizeof(unsigned);  // 96 KiB
    slstm_main<<<nblocks, 512, lds_bytes, stream>>>(h_t, x, h_sl, h_in, c_t,
                                                    bs, wp, outH, outC);
}
